// EPLSTMCell_49100066128158
// MI455X (gfx1250) — compile-verified, matrix operand order checked by reading, not run on hardware
//
#include <hip/hip_runtime.h>
#include <hip/hip_bf16.h>

// B=64, T=1024, D=256, H=256
#define BB 64
#define TT 1024
#define DD 256
#define HH 256
#define NCOL 1024   // 4*H gate columns

typedef __attribute__((ext_vector_type(16))) __bf16 v16bf;
typedef __attribute__((ext_vector_type(8)))  float  v8f;

__device__ inline v8f v8f_zero() {
  v8f z = {0.f,0.f,0.f,0.f,0.f,0.f,0.f,0.f};
  return z;
}

// ---- WMMA bf16 16x16x32 fragment loader (ISA 7.12.2 16-bit layout) ----
// Works for A (16x32 row-major) AND for B when B is stored transposed
// (N-major, B_T[n][k]): lane L holds row/column (L&15); k-offset +8 for
// lanes 16-31; elements 0..7 -> K=kb..kb+7, 8..15 -> K=kb+16..kb+23.
// Both 8-element runs are 16B contiguous -> two b128 loads per lane.
__device__ inline v16bf load_frag(const __bf16* base, int ld) {
  const int lane = threadIdx.x & 31;
  const int row  = lane & 15;
  const int kb   = (lane >> 4) ? 8 : 0;
  const __bf16* p = base + (size_t)row * ld + kb;
  v16bf f;
#pragma unroll
  for (int i = 0; i < 8; ++i) { f[i] = p[i]; f[i + 8] = p[16 + i]; }
  return f;
}

__device__ inline v8f wmma_bf16(v16bf a, v16bf b, v8f c) {
  return __builtin_amdgcn_wmma_f32_16x16x32_bf16(false, a, false, b,
                                                 (short)0, c, false, false);
}

// ---------------------------------------------------------------------------
// Kernel 0: x_seq [b][t][d] fp32 -> Xbf [t*B+b][d] bf16 (t-major tokens)
// ---------------------------------------------------------------------------
__global__ void k_conv_x(const float* __restrict__ x, __bf16* __restrict__ Xbf) {
  size_t i = (size_t)blockIdx.x * blockDim.x + threadIdx.x;   // [t][b][d]
  int d = (int)(i & 255);
  size_t tb = i >> 8;
  int b = (int)(tb & 63);
  size_t t = tb >> 6;
  Xbf[i] = (__bf16)x[((size_t)b * TT + t) * DD + d];
}

// ---------------------------------------------------------------------------
// Kernel 1: weight prep (tiny).  All B operands stored TRANSPOSED (N-major)
// so WMMA B-fragments load as two b128 per lane.
//  WcT[g*H+j][d] = Wx[g][d][j] + sum_h W_in[d][h]*We[g][h][j]    (bf16)
//  WrT[g*H+j][h] = Wh[g][h][j] - We[g][h][j]                     (bf16)
//  WoT[d][h]     = W_out[h][d]                                   (bf16)
// ---------------------------------------------------------------------------
__global__ void k_prep_w(const float* __restrict__ W_in, const float* __restrict__ Wx,
                         const float* __restrict__ Wh,  const float* __restrict__ We,
                         const float* __restrict__ W_out,
                         __bf16* __restrict__ WcT, __bf16* __restrict__ WrT,
                         __bf16* __restrict__ WoT) {
  int i = blockIdx.x * blockDim.x + threadIdx.x;      // 0..262143
  if (i < DD * NCOL) {
    int col = i & (NCOL - 1);   // gate column g*H+j
    int d   = i >> 10;          // also reused as 'h' for WrT
    int g   = col >> 8;
    int j   = col & 255;
    float acc = Wx[((size_t)g * DD + d) * HH + j];
    const float* wrow  = W_in + (size_t)d * HH;
    const float* wecol = We + ((size_t)g * HH) * HH + j;
#pragma unroll 4
    for (int h = 0; h < HH; ++h) acc += wrow[h] * wecol[(size_t)h * HH];
    WcT[(size_t)col * DD + d] = (__bf16)acc;
    WrT[(size_t)col * HH + d] = (__bf16)(Wh[((size_t)g * HH + d) * HH + j] -
                                         We[((size_t)g * HH + d) * HH + j]);
  }
  if (i < HH * DD) {
    int d2 = i & 255;
    int h2 = i >> 8;
    WoT[(size_t)d2 * HH + h2] = (__bf16)W_out[i];
  }
}

// ---------------------------------------------------------------------------
// Kernel 2: pre_static = Xbf @ Wc   (M=65536, N=1024, K=256) fp32 out.
// Per wave: 64x32 output tile (4 M-subtiles x 2 N-subtiles).
// ---------------------------------------------------------------------------
__global__ __launch_bounds__(1024) void k_gemm_pre(const __bf16* __restrict__ X,
                                                   const __bf16* __restrict__ WcT,
                                                   float* __restrict__ pre) {
  const int w  = threadIdx.x >> 5;
  const int gw = blockIdx.x * 32 + w;
  const int nbase = (gw & 31) * 32;     // 32 N-groups of 32 cols
  const int mbase = (gw >> 5) * 64;     // 1024 M-blocks of 64 rows
  v8f acc[4][2];
#pragma unroll
  for (int s = 0; s < 4; ++s)
#pragma unroll
    for (int n2 = 0; n2 < 2; ++n2) acc[s][n2] = v8f_zero();

  for (int kk = 0; kk < DD; kk += 32) {
    v16bf bf[2];
#pragma unroll
    for (int n2 = 0; n2 < 2; ++n2)
      bf[n2] = load_frag(WcT + (size_t)(nbase + n2 * 16) * DD + kk, DD);
#pragma unroll
    for (int s = 0; s < 4; ++s) {
      v16bf af = load_frag(X + (size_t)(mbase + s * 16) * DD + kk, DD);
#pragma unroll
      for (int n2 = 0; n2 < 2; ++n2)
        acc[s][n2] = wmma_bf16(af, bf[n2], acc[s][n2]);
    }
  }
  const int lane = threadIdx.x & 31;
  const int rhi  = (lane >> 4) ? 8 : 0;
#pragma unroll
  for (int s = 0; s < 4; ++s)
#pragma unroll
    for (int n2 = 0; n2 < 2; ++n2) {
      const int col = nbase + n2 * 16 + (lane & 15);
#pragma unroll
      for (int r = 0; r < 8; ++r)
        pre[(size_t)(mbase + s * 16 + r + rhi) * NCOL + col] = acc[s][n2][r];
    }
}

// ---------------------------------------------------------------------------
// Kernel 3: sequential scan.  One workgroup (32 waves), h in LDS (bf16),
// c in registers.  Per step: pre = pre_static[t] + h @ Wr + bias; gates;
// write h_seq/c_seq fp32 + h bf16 for the output GEMM.
// ---------------------------------------------------------------------------
#define HPAD 264   // LDS row stride in bf16 (256 + 8 pad -> conflict-free rows)

__global__ __launch_bounds__(1024) void k_scan(const float* __restrict__ pre,
                                               const __bf16* __restrict__ WrT,
                                               const float* __restrict__ bias,
                                               const float* __restrict__ h0,
                                               const float* __restrict__ c0,
                                               float* __restrict__ h_seq,
                                               float* __restrict__ c_seq,
                                               __bf16* __restrict__ hbf) {
  __shared__ __bf16 hlds[BB][HPAD];
  const int tid  = threadIdx.x;
  const int w    = tid >> 5;
  const int lane = tid & 31;

  // init h LDS from h_prev
  for (int i = tid; i < BB * HH; i += 1024)
    hlds[i >> 8][i & 255] = (__bf16)h0[i];

  // this wave's two 16x16 h-tiles (consecutive -> same batch-row block)
  const int ht0   = w * 2;
  const int mbase = (ht0 >> 4) * 16;                 // batch-row base
  const int nb[2] = { (ht0 & 15) * 16, ((ht0 + 1) & 15) * 16 };
  const int lrow  = lane & 15;
  const int hi8   = (lane >> 4) ? 8 : 0;

  v8f   cst[2];
  float bia[2][4];
#pragma unroll
  for (int t2 = 0; t2 < 2; ++t2) {
#pragma unroll
    for (int r = 0; r < 8; ++r)
      cst[t2][r] = c0[(size_t)(mbase + r + hi8) * HH + nb[t2] + lrow];
#pragma unroll
    for (int g = 0; g < 4; ++g)
      bia[t2][g] = bias[g * HH + nb[t2] + lrow];
  }
  __syncthreads();

  for (int t = 0; t < TT; ++t) {
    v8f acc[2][4];
#pragma unroll
    for (int t2 = 0; t2 < 2; ++t2)
#pragma unroll
      for (int g = 0; g < 4; ++g) acc[t2][g] = v8f_zero();

    for (int kk = 0; kk < HH; kk += 32) {
      v16bf af = load_frag(&hlds[mbase][0] + kk, HPAD);
#pragma unroll
      for (int t2 = 0; t2 < 2; ++t2)
#pragma unroll
        for (int g = 0; g < 4; ++g) {
          v16bf bf = load_frag(WrT + (size_t)(g * HH + nb[t2]) * HH + kk, HH);
          acc[t2][g] = wmma_bf16(af, bf, acc[t2][g]);
        }
    }
    __syncthreads();   // everyone finished reading h(t) from LDS

    const float* pret = pre + (size_t)t * BB * NCOL;
#pragma unroll
    for (int t2 = 0; t2 < 2; ++t2) {
      const int colh = nb[t2] + lrow;
#pragma unroll
      for (int r = 0; r < 8; ++r) {
        const int brow = mbase + r + hi8;
        const float* pr = pret + (size_t)brow * NCOL + colh;
        float pf = acc[t2][0][r] + pr[0 * HH] + bia[t2][0];
        float pi = acc[t2][1][r] + pr[1 * HH] + bia[t2][1];
        float po = acc[t2][2][r] + pr[2 * HH] + bia[t2][2];
        float pg = acc[t2][3][r] + pr[3 * HH] + bia[t2][3];
        float f  = 1.f / (1.f + __expf(-pf));
        float ii = 1.f / (1.f + __expf(-pi));
        float oo = 1.f / (1.f + __expf(-po));
        float gg = tanhf(pg);
        float cn = f * cst[t2][r] + ii * gg;
        float hn = oo * tanhf(cn);
        cst[t2][r] = cn;
        hlds[brow][colh] = (__bf16)hn;
        size_t o = ((size_t)brow * TT + t) * HH + colh;  // [b][t][h]
        h_seq[o] = hn;
        c_seq[o] = cn;
        hbf[o]   = (__bf16)hn;
      }
    }
    __syncthreads();   // h(t+1) visible in LDS for next step
  }
}

// ---------------------------------------------------------------------------
// Kernel 4: y = Hbf @ Wo + b_out   (M=65536 tokens [b*T+t], N=256, K=256)
// Per wave: 64x32 output tile.
// ---------------------------------------------------------------------------
__global__ __launch_bounds__(1024) void k_gemm_out(const __bf16* __restrict__ Hbf,
                                                   const __bf16* __restrict__ WoT,
                                                   const float* __restrict__ bout,
                                                   float* __restrict__ y) {
  const int w  = threadIdx.x >> 5;
  const int gw = blockIdx.x * 32 + w;
  const int nbase = (gw & 7) * 32;      // 8 N-groups of 32 cols
  const int mbase = (gw >> 3) * 64;     // 1024 M-blocks of 64 rows
  v8f acc[4][2];
#pragma unroll
  for (int s = 0; s < 4; ++s)
#pragma unroll
    for (int n2 = 0; n2 < 2; ++n2) acc[s][n2] = v8f_zero();

  for (int kk = 0; kk < HH; kk += 32) {
    v16bf bf[2];
#pragma unroll
    for (int n2 = 0; n2 < 2; ++n2)
      bf[n2] = load_frag(WoT + (size_t)(nbase + n2 * 16) * HH + kk, HH);
#pragma unroll
    for (int s = 0; s < 4; ++s) {
      v16bf af = load_frag(Hbf + (size_t)(mbase + s * 16) * HH + kk, HH);
#pragma unroll
      for (int n2 = 0; n2 < 2; ++n2)
        acc[s][n2] = wmma_bf16(af, bf[n2], acc[s][n2]);
    }
  }
  const int lane = threadIdx.x & 31;
  const int rhi  = (lane >> 4) ? 8 : 0;
#pragma unroll
  for (int s = 0; s < 4; ++s)
#pragma unroll
    for (int n2 = 0; n2 < 2; ++n2) {
      const int col = nbase + n2 * 16 + (lane & 15);
      const float bb = bout[col];
#pragma unroll
      for (int r = 0; r < 8; ++r)
        y[(size_t)(mbase + s * 16 + r + rhi) * DD + col] = acc[s][n2][r] + bb;
    }
}

// ---------------------------------------------------------------------------
extern "C" void kernel_launch(void* const* d_in, const int* in_sizes, int n_in,
                              void* d_out, int out_size, void* d_ws, size_t ws_size,
                              hipStream_t stream) {
  const float* x_seq  = (const float*)d_in[0];
  const float* h_prev = (const float*)d_in[1];
  const float* c_prev = (const float*)d_in[2];
  const float* W_in   = (const float*)d_in[3];
  const float* Wx     = (const float*)d_in[4];
  const float* Wh     = (const float*)d_in[5];
  const float* We     = (const float*)d_in[6];
  const float* bias   = (const float*)d_in[7];
  const float* W_out  = (const float*)d_in[8];
  const float* b_out  = (const float*)d_in[9];

  float* y     = (float*)d_out;                       // [B,T,D]
  float* h_seq = y + (size_t)BB * TT * HH;            // [B,T,H]
  float* c_seq = h_seq + (size_t)BB * TT * HH;        // [B,T,H]

  // workspace carve-up (bytes, 256B aligned)
  char* ws = (char*)d_ws;
  size_t off = 0;
  auto carve = [&](size_t bytes) {
    char* p = ws + off;
    off = (off + bytes + 255) & ~(size_t)255;
    return p;
  };
  __bf16* Xbf = (__bf16*)carve((size_t)BB * TT * DD * 2);       // 32 MB
  __bf16* WcT = (__bf16*)carve((size_t)DD * NCOL * 2);          // 512 KB
  __bf16* WrT = (__bf16*)carve((size_t)HH * NCOL * 2);          // 512 KB
  __bf16* WoT = (__bf16*)carve((size_t)HH * DD * 2);            // 128 KB
  float*  pre = (float*)carve((size_t)BB * TT * NCOL * 4);      // 256 MB
  __bf16* Hbf = (__bf16*)carve((size_t)BB * TT * HH * 2);       // 32 MB
  (void)ws_size; (void)in_sizes; (void)n_in; (void)out_size;

  k_conv_x<<<(BB * TT * DD) / 256, 256, 0, stream>>>(x_seq, Xbf);
  k_prep_w<<<(DD * NCOL) / 256, 256, 0, stream>>>(W_in, Wx, Wh, We, W_out,
                                                  WcT, WrT, WoT);
  k_gemm_pre<<<1024, 1024, 0, stream>>>(Xbf, WcT, pre);
  k_scan<<<1, 1024, 0, stream>>>(pre, WrT, bias, h_prev, c_prev, h_seq, c_seq, Hbf);
  k_gemm_out<<<256, 1024, 0, stream>>>(Hbf, WoT, b_out, y);
}
